// RotaryPEMultiHeadAttention_80728205295842
// MI455X (gfx1250) — matrix operand order repaired and checked by reading, not compiled, not run on hardware
//
#include <hip/hip_runtime.h>
#include <hip/hip_bf16.h>

typedef __attribute__((ext_vector_type(16))) _Float16 v16h;
typedef __attribute__((ext_vector_type(8)))  _Float16 v8h;
typedef __attribute__((ext_vector_type(4)))  _Float16 v4h;
typedef __attribute__((ext_vector_type(8)))  float    v8f;
typedef __attribute__((ext_vector_type(4)))  float    v4f;

static constexpr int D_MODEL = 1024;
static constexpr int HEADS   = 16;
static constexpr int DK      = 64;
static constexpr int BATCH   = 4;
static constexpr int SEQ     = 2048;
static constexpr int MTOT    = BATCH * SEQ;   // 8192

// ---------------------------------------------------------------- fragments
__device__ __forceinline__ v16h cat16(v8h lo, v8h hi) {
  return __builtin_shufflevector(lo, hi, 0,1,2,3,4,5,6,7,8,9,10,11,12,13,14,15);
}
// A-operand (16x32 f16): lanes 0-15 hold K{0..7,16..23}, lanes 16-31 K{8..15,24..31}.
// p = row pointer, contiguous in K. sel = lane>=16.
__device__ __forceinline__ v16h fragA(const _Float16* p, int sel) {
  v8h lo = *(const v8h*)(p + sel * 8);
  v8h hi = *(const v8h*)(p + 16 + sel * 8);
  return cat16(lo, hi);
}
// B-operand (32x16 f16): lanes 0-15 hold K{0..7,16..23}, lanes 16-31 K{8..15,24..31} of col n=lane%16.
// p = column pointer, contiguous in K. sel = lane>=16.
__device__ __forceinline__ v16h fragB(const _Float16* p, int sel) {
  v8h lo = *(const v8h*)(p + sel * 8);
  v8h hi = *(const v8h*)(p + 16 + sel * 8);
  return cat16(lo, hi);
}
__device__ __forceinline__ v8f wmma_f16(v16h a, v16h b, v8f c) {
  return __builtin_amdgcn_wmma_f32_16x16x32_f16(false, a, false, b, (short)0, c,
                                                false, false);
}
// CDNA5 async global->LDS 16-byte copy (ASYNCcnt-tracked, no VGPR data round-trip).
__device__ __forceinline__ void async_copy16(_Float16* ldst, const _Float16* gsrc) {
  unsigned l = (unsigned)(uintptr_t)ldst;              // generic addr low 32b == LDS byte offset
  unsigned long long g = (unsigned long long)(uintptr_t)gsrc;
  asm volatile("global_load_async_to_lds_b128 %0, %1, off"
               :: "v"(l), "v"(g) : "memory");
}
__device__ __forceinline__ void async_wait0() {
  asm volatile("s_wait_asynccnt 0x0" ::: "memory");
}

// ---------------------------------------------------------------- GEMM
// C[M,N] = A[M,K](f32) * W[K,N](f32) + bias[N], f16 WMMA, f32 accumulate.
// M=8192, N=K=1024. Block tile 128x128, 8 waves of 32x64, BK=64 (2 K-steps/stage).
static constexpr int BM = 128, BN = 128, BKK = 64, LDA = 72, LDB = 72;

__global__ __launch_bounds__(256)
void gemm_kernel(const float* __restrict__ A, const float* __restrict__ W,
                 const float* __restrict__ bias, float* __restrict__ C)
{
  constexpr int N = D_MODEL, K = D_MODEL;
  __shared__ alignas(16) _Float16 sA[BM * LDA];   // [m][k], padded (18 KB)
  __shared__ alignas(16) _Float16 sB[BN * LDB];   // [n][k] (W transposed), padded (18 KB)
  const int tid  = threadIdx.x;
  const int lane = tid & 31;
  const int wave = tid >> 5;
  const int l16  = lane & 15;
  const int sel  = lane >> 4;
  const int bm = blockIdx.x * BM;
  const int bn = blockIdx.y * BN;
  const int wm = (wave & 3) * 32;   // 4 waves along M
  const int wn = (wave >> 2) * 64;  // 2 waves along N

  v8f acc[2][4];
#pragma unroll
  for (int i = 0; i < 2; ++i)
#pragma unroll
    for (int j = 0; j < 4; ++j)
      acc[i][j] = v8f{0.f,0.f,0.f,0.f,0.f,0.f,0.f,0.f};

  for (int k0 = 0; k0 < K; k0 += BKK) {
    // stage A tile 128x64 (f32 -> f16): 8 passes of 16 rows
#pragma unroll
    for (int p = 0; p < 8; ++p) {
      int r = p * 16 + (tid >> 4);
      int c = (tid & 15) * 4;
      const float* g = A + (size_t)(bm + r) * K + k0 + c;
      if (p == 0 && k0 + BKK < K) __builtin_prefetch(g + BKK, 0, 3);
      v4f f = *(const v4f*)g;
      v4h h = { (_Float16)f.x, (_Float16)f.y, (_Float16)f.z, (_Float16)f.w };
      *(v4h*)(sA + r * LDA + c) = h;
    }
    // stage W tile 64x128, transposed into sB[n][k]: 8 passes of 8 k-rows
#pragma unroll
    for (int p = 0; p < 8; ++p) {
      int kk = p * 8 + (tid >> 5);
      int n  = (tid & 31) * 4;
      const float* g = W + (size_t)(k0 + kk) * N + bn + n;
      if (p == 0 && k0 + BKK < K) __builtin_prefetch(g + (size_t)BKK * N, 0, 3);
      v4f f = *(const v4f*)g;
      sB[(n + 0) * LDB + kk] = (_Float16)f.x;
      sB[(n + 1) * LDB + kk] = (_Float16)f.y;
      sB[(n + 2) * LDB + kk] = (_Float16)f.z;
      sB[(n + 3) * LDB + kk] = (_Float16)f.w;
    }
    __syncthreads();
    // two K=32 WMMA steps per staged tile -> 16 WMMAs between barriers
#pragma unroll
    for (int ks = 0; ks < 2; ++ks) {
      v16h af[2], bf[4];
#pragma unroll
      for (int i = 0; i < 2; ++i)
        af[i] = fragA(sA + (wm + i * 16 + l16) * LDA + ks * 32, sel);
#pragma unroll
      for (int j = 0; j < 4; ++j)
        bf[j] = fragB(sB + (wn + j * 16 + l16) * LDB + ks * 32, sel);
#pragma unroll
      for (int i = 0; i < 2; ++i)
#pragma unroll
        for (int j = 0; j < 4; ++j)
          acc[i][j] = wmma_f16(af[i], bf[j], acc[i][j]);
    }
    __syncthreads();
  }
  // epilogue: C/D layout -> lane n = l16, VGPR r = row r + 8*sel
#pragma unroll
  for (int i = 0; i < 2; ++i)
#pragma unroll
    for (int j = 0; j < 4; ++j) {
      int n = bn + wn + j * 16 + l16;
      float bv = bias[n];
#pragma unroll
      for (int r = 0; r < 8; ++r) {
        int m = bm + wm + i * 16 + sel * 8 + r;
        C[(size_t)m * N + n] = acc[i][j][r] + bv;
      }
    }
}

// ---------------------------------------------------------------- RoPE + pack
// mode 0: rope first 32 dims, pack f16 [B,H,S,64]
// mode 1: no rope, pack f16 transposed [B,H,64,S]  (for V)
__global__ __launch_bounds__(256)
void rope_pack_kernel(const float* __restrict__ X, _Float16* __restrict__ out, int mode)
{
  size_t idx = (size_t)blockIdx.x * 256 + threadIdx.x;
  int d = (int)(idx & 63);
  int h = (int)((idx >> 6) & 15);
  size_t bs = idx >> 10;
  int s = (int)(bs % SEQ);
  int b = (int)(bs / SEQ);
  float v = X[idx];
  float o = v;
  if (mode == 0 && d < 32) {
    int i = d & 15;
    float theta = __powf(10000.0f, -(float)i * (1.0f / 16.0f));
    float ang = (float)s * theta;
    float cs = __cosf(ang), sn = __sinf(ang);
    float other = X[idx + ((d < 16) ? 16 : -16)];
    o = (d < 16) ? (v * cs - other * sn) : (v * cs + other * sn);
  }
  size_t dst;
  if (mode == 1) dst = (((size_t)b * HEADS + h) * DK + d) * SEQ + s;
  else           dst = (((size_t)b * HEADS + h) * SEQ + s) * DK + d;
  out[dst] = (_Float16)o;
}

// ---------------------------------------------------------------- attention
// Q,K: f16 [B,H,S,64]; Vt: f16 [B,H,64,S]; X out f32 [B,S,1024].
// Block = 8 waves * 16 query rows = 128 rows of one (b,h). Flash over S keys.
// K/V tiles staged with CDNA5 async global->LDS copies (ASYNCcnt).
static constexpr int LKD = 72;  // sK row stride (halves)
static constexpr int LVD = 40;  // sV row stride
static constexpr int LPD = 40;  // sP row stride

__global__ __launch_bounds__(256)
void attn_kernel(const _Float16* __restrict__ Q, const _Float16* __restrict__ Kp,
                 const _Float16* __restrict__ Vt, float* __restrict__ X)
{
  __shared__ alignas(16) _Float16 sK[32 * LKD];       // [j][d]
  __shared__ alignas(16) _Float16 sV[64 * LVD];       // [d][j]
  __shared__ alignas(16) _Float16 sP[8 * 16 * LPD];   // per-wave 16x32 P tile
  const int tid  = threadIdx.x;
  const int lane = tid & 31;
  const int wave = tid >> 5;
  const int l16  = lane & 15;
  const int sel  = lane >> 4;
  const int bh   = blockIdx.y;            // b*16 + h
  const size_t base = (size_t)bh * SEQ * DK;
  const int i0 = blockIdx.x * 128 + wave * 16;

  // Q fragments for this wave's 16 rows, kept in registers for whole pass
  const _Float16* qrow = Q + base + (size_t)(i0 + l16) * DK;
  v16h aQ0 = fragA(qrow, sel);        // d 0..31
  v16h aQ1 = fragA(qrow + 32, sel);   // d 32..63

  v8f O[4];
  float mrow[8], lrow[8];
#pragma unroll
  for (int t = 0; t < 4; ++t) O[t] = v8f{0.f,0.f,0.f,0.f,0.f,0.f,0.f,0.f};
#pragma unroll
  for (int r = 0; r < 8; ++r) { mrow[r] = -1e30f; lrow[r] = 0.f; }

  _Float16* myP = sP + wave * 16 * LPD;
  const v8f zf = v8f{0.f,0.f,0.f,0.f,0.f,0.f,0.f,0.f};
  const float scale = 0.125f;  // 1/sqrt(64)

  // staging addresses are loop-invariant except for the j0 offset
  const int kr = tid >> 3, kc = (tid & 7) * 8;   // K: 32 rows x 64 halves
  const int vr = tid >> 2, vc = (tid & 3) * 8;   // V^T: 64 rows x 32 halves

  for (int j0 = 0; j0 < SEQ; j0 += 32) {
    __syncthreads();
    async_copy16(sK + kr * LKD + kc, Kp + base + (size_t)(j0 + kr) * DK + kc);
    async_copy16(sV + vr * LVD + vc, Vt + base + (size_t)vr * SEQ + j0 + vc);
    async_wait0();
    __syncthreads();

    // scores: two 16x16 j-tiles, K-contraction = 64 (2 WMMAs each)
    v8f s0 = wmma_f16(aQ0, fragB(sK + l16 * LKD, sel), zf);
    s0     = wmma_f16(aQ1, fragB(sK + l16 * LKD + 32, sel), s0);
    v8f s1 = wmma_f16(aQ0, fragB(sK + (16 + l16) * LKD, sel), zf);
    s1     = wmma_f16(aQ1, fragB(sK + (16 + l16) * LKD + 32, sel), s1);

    // online softmax per row (rows r+8*sel live across 16-lane halves)
#pragma unroll
    for (int r = 0; r < 8; ++r) {
      float v0 = s0[r] * scale, v1 = s1[r] * scale;
      float mx = fmaxf(v0, v1);
#pragma unroll
      for (int off = 8; off >= 1; off >>= 1)
        mx = fmaxf(mx, __shfl_xor(mx, off, 32));
      float mnew = fmaxf(mrow[r], mx);
      float corr = __expf(mrow[r] - mnew);
      float p0 = __expf(v0 - mnew);
      float p1 = __expf(v1 - mnew);
      float ps = p0 + p1;
#pragma unroll
      for (int off = 8; off >= 1; off >>= 1)
        ps += __shfl_xor(ps, off, 32);
      lrow[r] = lrow[r] * corr + ps;
      mrow[r] = mnew;
#pragma unroll
      for (int t = 0; t < 4; ++t) O[t][r] *= corr;
      int row = r + sel * 8;
      myP[row * LPD + l16]      = (_Float16)p0;   // j-tile 0
      myP[row * LPD + 16 + l16] = (_Float16)p1;   // j-tile 1
    }
    // P (C-layout) -> A-layout via wave-private LDS tile; DS ops are in-order per wave
    v16h aP = fragA(myP + l16 * LPD, sel);
#pragma unroll
    for (int t = 0; t < 4; ++t)
      O[t] = wmma_f16(aP, fragB(sV + (t * 16 + l16) * LVD, sel), O[t]);
  }

  const int b = bh >> 4, h = bh & 15;
#pragma unroll
  for (int t = 0; t < 4; ++t)
#pragma unroll
    for (int r = 0; r < 8; ++r) {
      int row = i0 + sel * 8 + r;
      X[((size_t)b * SEQ + row) * D_MODEL + h * DK + t * 16 + l16] =
          O[t][r] / lrow[r];
    }
}

// ---------------------------------------------------------------- launch
extern "C" void kernel_launch(void* const* d_in, const int* in_sizes, int n_in,
                              void* d_out, int out_size, void* d_ws, size_t ws_size,
                              hipStream_t stream) {
  (void)in_sizes; (void)n_in; (void)out_size; (void)ws_size;
  const float* query = (const float*)d_in[0];
  const float* key   = (const float*)d_in[1];
  const float* value = (const float*)d_in[2];
  const float* Wq = (const float*)d_in[3];
  const float* bq = (const float*)d_in[4];
  const float* Wk = (const float*)d_in[5];
  const float* bk = (const float*)d_in[6];
  const float* Wv = (const float*)d_in[7];
  const float* bv = (const float*)d_in[8];
  const float* Wo = (const float*)d_in[9];
  const float* bo = (const float*)d_in[10];
  float* out = (float*)d_out;

  char* ws = (char*)d_ws;
  float* r0 = (float*)ws;                               // 8192*1024 f32 scratch
  size_t R0 = (size_t)MTOT * D_MODEL * sizeof(float);
  _Float16* qh = (_Float16*)(ws + R0);
  _Float16* kh = qh + (size_t)MTOT * D_MODEL;
  _Float16* vh = kh + (size_t)MTOT * D_MODEL;

  dim3 tb(256);
  dim3 gG(MTOT / BM, D_MODEL / BN);                     // 64 x 8
  int gR = (MTOT * D_MODEL) / 256;                      // 32768
  dim3 gA(SEQ / 128, BATCH * HEADS);                    // 16 x 64

  gemm_kernel<<<gG, tb, 0, stream>>>(query, Wq, bq, r0);
  rope_pack_kernel<<<gR, tb, 0, stream>>>(r0, qh, 0);
  gemm_kernel<<<gG, tb, 0, stream>>>(key, Wk, bk, r0);
  rope_pack_kernel<<<gR, tb, 0, stream>>>(r0, kh, 0);
  gemm_kernel<<<gG, tb, 0, stream>>>(value, Wv, bv, r0);
  rope_pack_kernel<<<gR, tb, 0, stream>>>(r0, vh, 1);
  attn_kernel<<<gA, tb, 0, stream>>>(qh, kh, vh, r0);
  gemm_kernel<<<gG, tb, 0, stream>>>(r0, Wo, bo, out);
}
